// SelectMolAttachment_72653666779410
// MI455X (gfx1250) — matrix operand order repaired and checked by reading, not compiled, not run on hardware
//
#include <hip/hip_runtime.h>
#include <stdint.h>
#include <stddef.h>

// ---------------- types ----------------
typedef __bf16 bf16;
typedef __attribute__((ext_vector_type(16))) __bf16 v16bf;
typedef __attribute__((ext_vector_type(8)))  __bf16 v8bf;
typedef __attribute__((ext_vector_type(8)))  float  v8f;

union V16U { v16bf v; v8bf h[2]; };

#define BATCH    4096
#define MAXA     50
#define PADM     64
#define RTOT     (BATCH * PADM)     // 262144 padded rows
#define NODE_DIM 256
#define ATTN_DIM 128
#define REPR_DIM 512

// ---------------------------------------------------------------------
// CDNA5 async global->LDS copy (tracked by ASYNCcnt).
// The LDS destination VGPR takes the wave-relative LDS byte address;
// per ISA 10.2 the low 32 bits of a generic pointer to __shared__ data
// are exactly that offset.
// ---------------------------------------------------------------------
static __device__ __forceinline__ void async_b128(void* lds_dst, const void* gsrc)
{
    uint32_t l = (uint32_t)(uintptr_t)lds_dst;
    asm volatile("global_load_async_to_lds_b128 %0, %1, off"
                 :: "v"(l), "v"(gsrc)
                 : "memory");
}
static __device__ __forceinline__ void wait_async0()
{
    asm volatile("s_wait_asynccnt 0x0" ::: "memory");
}

// =====================================================================
// f32 -> bf16 conversion (weights, small)
// =====================================================================
__global__ __launch_bounds__(256) void f32_to_bf16_kernel(
    const float* __restrict__ src, bf16* __restrict__ dst, int n)
{
    int i = blockIdx.x * blockDim.x + threadIdx.x;
    if (i < n) dst[i] = (bf16)src[i];
}

// =====================================================================
// ragged gather -> padded [RTOT, 256] bf16, zeros for invalid rows
// =====================================================================
__global__ __launch_bounds__(256) void gather_pad_kernel(
    const float* __restrict__ nh, const int* __restrict__ counts,
    const int* __restrict__ offs, bf16* __restrict__ inp)
{
    size_t t = (size_t)blockIdx.x * blockDim.x + threadIdx.x;
    int    c   = (int)(t & 31) * 8;      // 0..248
    size_t row = t >> 5;                 // 0..RTOT-1
    int    b   = (int)(row >> 6);
    int    p   = (int)(row & 63);
    bf16* dst = inp + row * NODE_DIM + c;
    if (p < counts[b]) {
        const float* src = nh + ((size_t)offs[b] + p) * NODE_DIM + c;
        #pragma unroll
        for (int i = 0; i < 8; ++i) dst[i] = (bf16)src[i];
    } else {
        #pragma unroll
        for (int i = 0; i < 8; ++i) dst[i] = (bf16)0.f;
    }
}

// =====================================================================
// concat(dst_repr, target_repr) -> bf16 [4096, 1024]
// =====================================================================
__global__ __launch_bounds__(256) void concat_kernel(
    const float* __restrict__ a, const float* __restrict__ b,
    bf16* __restrict__ o)
{
    int t   = blockIdx.x * blockDim.x + threadIdx.x;   // BATCH*128 threads
    int row = t >> 7;
    int c   = (t & 127) * 8;
    const float* src = (c < REPR_DIM) ? (a + (size_t)row * REPR_DIM + c)
                                      : (b + (size_t)row * REPR_DIM + (c - REPR_DIM));
    bf16* dst = o + (size_t)row * (2 * REPR_DIM) + c;
    #pragma unroll
    for (int i = 0; i < 8; ++i) dst[i] = (bf16)src[i];
}

// =====================================================================
// Generic bf16 GEMM:  C[M,N] = act(A[M,K] @ W[K,N] + bias)
// block = 256 threads (8 waves), tile 128(M) x 64(N), K-step 32
// A tile staged via async global->LDS; W tile transposed via VGPRs.
// requires: M % 128 == 0, N % 64 == 0, K % 32 == 0
// =====================================================================
__global__ __launch_bounds__(256) void gemm_bf16_kernel(
    const bf16* __restrict__ A, const bf16* __restrict__ W,
    const float* __restrict__ bias, bf16* __restrict__ C,
    int K, int N, int relu)
{
    __shared__ __align__(32) bf16 As[128][32];   // A tile, row-major
    __shared__ __align__(32) bf16 Wt[64][32];    // W tile, transposed (n-major)

    const int tid  = threadIdx.x;
    const int lane = tid & 31;
    const int wave = tid >> 5;
    const int wm   = wave & 3;        // 0..3  (M sub-tile)
    const int wn   = wave >> 2;       // 0..1  (N sub-tile)
    const size_t gm = (size_t)blockIdx.y * 128;
    const int    gn = blockIdx.x * 64;

    const v8f zero = {0.f,0.f,0.f,0.f,0.f,0.f,0.f,0.f};
    v8f acc[2][2];
    acc[0][0] = zero; acc[0][1] = zero; acc[1][0] = zero; acc[1][1] = zero;

    const int arow = tid >> 1;             // 0..127
    const int acol = (tid & 1) * 16;       // 0 or 16
    const int wkr  = tid >> 3;             // 0..31
    const int wcg  = (tid & 7) * 8;        // 0..56

    for (int k0 = 0; k0 < K; k0 += 32) {
        // ---- stage A tile 128x32 via async copies (2 x b128/thread) ----
        {
            const bf16* g = A + (gm + arow) * (size_t)K + k0 + acol;
            async_b128(&As[arow][acol],     g);
            async_b128(&As[arow][acol + 8], g + 8);
        }
        // ---- stage W tile transposed: 32x64 -> Wt[64][32] ----
        {
            v8bf w = *(const v8bf*)(W + (size_t)(k0 + wkr) * N + gn + wcg);
            #pragma unroll
            for (int i = 0; i < 8; ++i) Wt[wcg + i][wkr] = w[i];
        }
        // prefetch next W tile row into cache while WMMAs run
        if (k0 + 32 < K)
            __builtin_prefetch(W + (size_t)(k0 + 32 + wkr) * N + gn + wcg, 0, 1);

        wait_async0();
        __syncthreads();

        // ---- build fragments and WMMA ----
        v16bf afrag[2];
        #pragma unroll
        for (int mi = 0; mi < 2; ++mi) {
            int m  = wm * 32 + mi * 16 + (lane & 15);
            int kh = (lane < 16) ? 0 : 16;
            V16U u;
            u.h[0] = *(const v8bf*)(&As[m][kh]);
            u.h[1] = *(const v8bf*)(&As[m][kh + 8]);
            afrag[mi] = u.v;
        }
        v16bf bfrag[2];
        #pragma unroll
        for (int ni = 0; ni < 2; ++ni) {
            int c  = wn * 32 + ni * 16 + (lane & 15);
            int kb = (lane < 16) ? 0 : 16;
            bfrag[ni] = *(const v16bf*)(&Wt[c][kb]);
        }
        #pragma unroll
        for (int mi = 0; mi < 2; ++mi)
            #pragma unroll
            for (int ni = 0; ni < 2; ++ni)
                acc[mi][ni] = __builtin_amdgcn_wmma_f32_16x16x32_bf16(
                    false, afrag[mi], false, bfrag[ni],
                    (short)0, acc[mi][ni], false, false);
        __syncthreads();
    }

    // ---- epilogue: bias, relu, store bf16 ----
    #pragma unroll
    for (int mi = 0; mi < 2; ++mi) {
        #pragma unroll
        for (int ni = 0; ni < 2; ++ni) {
            int col   = gn + wn * 32 + ni * 16 + (lane & 15);
            float bv  = bias[col];
            size_t rb = gm + wm * 32 + mi * 16 + ((lane < 16) ? 0 : 8);
            #pragma unroll
            for (int r = 0; r < 8; ++r) {
                float v = acc[mi][ni][r] + bv;
                if (relu) v = fmaxf(v, 0.f);
                C[(rb + r) * (size_t)N + col] = (bf16)v;
            }
        }
    }
}

// =====================================================================
// Per-molecule attention: S = K@Q^T, masked softmax, corr = attn@V
// one block (256 thr, 8 waves) per molecule; 64x128 bf16 tiles in LDS
// =====================================================================
__global__ __launch_bounds__(256) void attn_kernel(
    const bf16* __restrict__ Kg, const bf16* __restrict__ Qg,
    const bf16* __restrict__ Vg, const int* __restrict__ counts,
    bf16* __restrict__ corr)
{
    __shared__ __align__(32) char sKS[64 * 128 * 2]; // Ks bf16[64][128] / Sf f32[64][64]
    __shared__ __align__(32) bf16 Qs[64][128];
    __shared__ __align__(32) bf16 Vt[128][64];       // V transposed (d-major)
    __shared__ __align__(32) bf16 Aw[64][64];        // attention weights bf16

    bf16  (*Ks)[128] = (bf16(*)[128])sKS;
    float (*Sf)[64]  = (float(*)[64])sKS;

    const int b    = blockIdx.x;
    const int tid  = threadIdx.x;
    const int lane = tid & 31;
    const int wave = tid >> 5;
    const int cnt  = counts[b];
    const size_t base = (size_t)b * PADM * ATTN_DIM;

    // ---- K, Q via async global->LDS; V transposed through VGPRs ----
    {
        int row = tid >> 2;           // 0..63
        int col = (tid & 3) * 32;     // 0,32,64,96
        const bf16* kg = Kg + base + (size_t)row * 128 + col;
        const bf16* qg = Qg + base + (size_t)row * 128 + col;
        #pragma unroll
        for (int i = 0; i < 4; ++i) {
            async_b128(&Ks[row][col + 8 * i], kg + 8 * i);
            async_b128(&Qs[row][col + 8 * i], qg + 8 * i);
        }
        const bf16* vs = Vg + base + (size_t)row * 128 + col;
        #pragma unroll
        for (int i = 0; i < 32; ++i) Vt[col + i][row] = vs[i];
    }
    wait_async0();
    __syncthreads();

    // ---- scores S = K @ Q^T : 16 tiles, 2 per wave ----
    const v8f zero = {0.f,0.f,0.f,0.f,0.f,0.f,0.f,0.f};
    v8f sacc[2]; sacc[0] = zero; sacc[1] = zero;
    #pragma unroll
    for (int i = 0; i < 2; ++i) {
        int t  = wave * 2 + i;
        int mt = t >> 2, nt = t & 3;
        #pragma unroll
        for (int d0 = 0; d0 < 128; d0 += 32) {
            int m  = mt * 16 + (lane & 15);
            int kh = (lane < 16) ? 0 : 16;
            V16U a;
            a.h[0] = *(const v8bf*)(&Ks[m][d0 + kh]);
            a.h[1] = *(const v8bf*)(&Ks[m][d0 + kh + 8]);
            int q  = nt * 16 + (lane & 15);
            int db = d0 + ((lane < 16) ? 0 : 16);
            v16bf bb = *(const v16bf*)(&Qs[q][db]);
            sacc[i] = __builtin_amdgcn_wmma_f32_16x16x32_bf16(
                false, a.v, false, bb, (short)0, sacc[i], false, false);
        }
    }
    __syncthreads();   // everyone done reading Ks

    // ---- spill scores f32 into Sf (aliases Ks) ----
    #pragma unroll
    for (int i = 0; i < 2; ++i) {
        int t  = wave * 2 + i;
        int mt = t >> 2, nt = t & 3;
        int q  = nt * 16 + (lane & 15);
        int rb = mt * 16 + ((lane < 16) ? 0 : 8);
        #pragma unroll
        for (int r = 0; r < 8; ++r) Sf[rb + r][q] = sacc[i][r];
    }
    __syncthreads();

    // ---- masked softmax per row (threads 0..63) ----
    if (tid < 64) {
        int r = tid;
        if (r >= MAXA) {
            for (int q = 0; q < 64; ++q) Aw[r][q] = (bf16)0.f;
        } else {
            const float NEG = -1e9f;
            const float ISD = 0.08838834764831845f;   // 1/sqrt(128)
            bool rm = (r >= cnt);
            float x[MAXA];
            float mx = -3.4e38f;
            for (int q = 0; q < MAXA; ++q) {
                float s = Sf[r][q];
                if (rm || q >= cnt) s = NEG;
                float v = s * ISD;
                x[q] = v;
                mx = fmaxf(mx, v);
            }
            float sum = 0.f;
            for (int q = 0; q < MAXA; ++q) { float e = __expf(x[q] - mx); x[q] = e; sum += e; }
            float inv = 1.f / sum;
            for (int q = 0; q < 64; ++q)
                Aw[r][q] = (bf16)((q < MAXA) ? x[q] * inv : 0.f);
        }
    }
    __syncthreads();

    // ---- corr = Aw @ V : 32 tiles, 4 per wave ----
    #pragma unroll
    for (int i = 0; i < 4; ++i) {
        int t  = wave * 4 + i;
        int mt = t >> 3, nt = t & 7;
        v8f cacc = zero;
        #pragma unroll
        for (int k0 = 0; k0 < 64; k0 += 32) {
            int m  = mt * 16 + (lane & 15);
            int kh = (lane < 16) ? 0 : 16;
            V16U a;
            a.h[0] = *(const v8bf*)(&Aw[m][k0 + kh]);
            a.h[1] = *(const v8bf*)(&Aw[m][k0 + kh + 8]);
            int d  = nt * 16 + (lane & 15);
            int qb = k0 + ((lane < 16) ? 0 : 16);
            v16bf bb = *(const v16bf*)(&Vt[d][qb]);
            cacc = __builtin_amdgcn_wmma_f32_16x16x32_bf16(
                false, a.v, false, bb, (short)0, cacc, false, false);
        }
        int d  = nt * 16 + (lane & 15);
        int rb = mt * 16 + ((lane < 16) ? 0 : 8);
        #pragma unroll
        for (int r = 0; r < 8; ++r)
            corr[base + (size_t)(rb + r) * 128 + d] = (bf16)cacc[r];
    }
}

// =====================================================================
// logits[b,k] = (K2[b,k,:] . Q2[b,:]) / sqrt(128)   (k < 50)
// =====================================================================
__global__ __launch_bounds__(256) void logits_kernel(
    const bf16* __restrict__ K2, const bf16* __restrict__ Q2,
    float* __restrict__ logits)
{
    int t = blockIdx.x * blockDim.x + threadIdx.x;  // BATCH*64
    int b = t >> 6, k = t & 63;
    if (k >= MAXA) return;
    const bf16* kr = K2 + (size_t)t * ATTN_DIM;
    const bf16* qr = Q2 + (size_t)b * ATTN_DIM;
    float acc = 0.f;
    #pragma unroll 8
    for (int i = 0; i < ATTN_DIM; ++i) acc += (float)kr[i] * (float)qr[i];
    logits[t] = acc * 0.08838834764831845f;
}

// =====================================================================
// sel MLP: [50]->relu(128)->[50], sigmoid.  one block per molecule
// =====================================================================
__global__ __launch_bounds__(128) void sel_kernel(
    const float* __restrict__ logits,  // [B*64]
    const float* __restrict__ W0, const float* __restrict__ b0,
    const float* __restrict__ W1, const float* __restrict__ b1,
    float* __restrict__ selected)      // [B*50]
{
    __shared__ float lg[MAXA];
    __shared__ float h[128];
    int b = blockIdx.x;
    int t = threadIdx.x;
    if (t < MAXA) lg[t] = logits[b * 64 + t];
    __syncthreads();
    float a = b0[t];
    for (int k = 0; k < MAXA; ++k) a += lg[k] * W0[k * 128 + t];
    h[t] = fmaxf(a, 0.f);
    __syncthreads();
    if (t < MAXA) {
        float o = b1[t];
        for (int j = 0; j < 128; ++j) o += h[j] * W1[j * MAXA + t];
        selected[b * MAXA + t] = 1.f / (1.f + __expf(-o));
    }
}

// =====================================================================
// scatter selected[B,50] -> ragged out[N]
// =====================================================================
__global__ __launch_bounds__(256) void scatter_kernel(
    const float* __restrict__ sel, const int* __restrict__ nb,
    const int* __restrict__ np, float* __restrict__ out, int N)
{
    int i = blockIdx.x * blockDim.x + threadIdx.x;
    if (i < N) out[i] = sel[nb[i] * MAXA + np[i]];
}

// =====================================================================
// host driver
// =====================================================================
extern "C" void kernel_launch(void* const* d_in, const int* in_sizes, int n_in,
                              void* d_out, int out_size, void* d_ws, size_t ws_size,
                              hipStream_t stream)
{
    (void)n_in; (void)out_size; (void)ws_size;
    const float* nh   = (const float*)d_in[0];
    const float* dstr = (const float*)d_in[1];
    const float* tgtr = (const float*)d_in[2];

    const float* wqW[3] = {(const float*)d_in[3],  (const float*)d_in[5],  (const float*)d_in[7]};
    const float* wqB[3] = {(const float*)d_in[4],  (const float*)d_in[6],  (const float*)d_in[8]};
    const float* wkW[3] = {(const float*)d_in[9],  (const float*)d_in[11], (const float*)d_in[13]};
    const float* wkB[3] = {(const float*)d_in[10], (const float*)d_in[12], (const float*)d_in[14]};
    const float* wvW[3] = {(const float*)d_in[15], (const float*)d_in[17], (const float*)d_in[19]};
    const float* wvB[3] = {(const float*)d_in[16], (const float*)d_in[18], (const float*)d_in[20]};
    const float* ukW[3] = {(const float*)d_in[21], (const float*)d_in[23], (const float*)d_in[25]};
    const float* ukB[3] = {(const float*)d_in[22], (const float*)d_in[24], (const float*)d_in[26]};
    const float* uqW[3] = {(const float*)d_in[27], (const float*)d_in[29], (const float*)d_in[31]};
    const float* uqB[3] = {(const float*)d_in[28], (const float*)d_in[30], (const float*)d_in[32]};
    const float* selW0 = (const float*)d_in[33];
    const float* selB0 = (const float*)d_in[34];
    const float* selW1 = (const float*)d_in[35];
    const float* selB1 = (const float*)d_in[36];
    const int* counts = (const int*)d_in[37];
    const int* offs   = (const int*)d_in[38];
    const int* nbat   = (const int*)d_in[39];
    const int* npos   = (const int*)d_in[40];
    const int Nnodes  = in_sizes[0] / NODE_DIM;

    // ---- carve workspace ----
    char* p = (char*)d_ws;
    auto take = [&](size_t bytes) {
        char* r = p;
        p += (bytes + 255) & ~(size_t)255;
        return r;
    };
    bf16* bufA = (bf16*)take((size_t)RTOT * 256 * 2);   // inp, later corr
    bf16* bufB = (bf16*)take((size_t)RTOT * 256 * 2);   // hidden ping
    bf16* bufC = (bf16*)take((size_t)RTOT * 256 * 2);   // hidden pong
    bf16* bufK = (bf16*)take((size_t)RTOT * 128 * 2);
    bf16* bufQ = (bf16*)take((size_t)RTOT * 128 * 2);   // later K2
    bf16* bufV = (bf16*)take((size_t)RTOT * 128 * 2);
    bf16* ucat = (bf16*)take((size_t)BATCH * 1024 * 2);
    bf16* u1   = (bf16*)take((size_t)BATCH * 384 * 2);
    bf16* u2   = (bf16*)take((size_t)BATCH * 256 * 2);
    bf16* q2   = (bf16*)take((size_t)BATCH * 128 * 2);
    float* logits = (float*)take((size_t)BATCH * 64 * 4);
    float* selbuf = (float*)take((size_t)BATCH * MAXA * 4);

    // ---- convert weight matrices to bf16 ----
    struct WSrc { const float* s; int n; };
    WSrc wl[15] = {
        {wqW[0],256*256},{wqW[1],256*256},{wqW[2],256*128},
        {wkW[0],256*256},{wkW[1],256*256},{wkW[2],256*128},
        {wvW[0],256*256},{wvW[1],256*256},{wvW[2],256*128},
        {ukW[0],128*256},{ukW[1],256*256},{ukW[2],256*128},
        {uqW[0],1024*384},{uqW[1],384*256},{uqW[2],256*128},
    };
    bf16* wbf[15];
    for (int i = 0; i < 15; ++i) {
        wbf[i] = (bf16*)take((size_t)wl[i].n * 2);
        f32_to_bf16_kernel<<<(wl[i].n + 255) / 256, 256, 0, stream>>>(wl[i].s, wbf[i], wl[i].n);
    }

    auto gemm = [&](const bf16* A, const bf16* W, const float* bias, bf16* C,
                    int M, int N, int K, int relu) {
        dim3 grid(N / 64, M / 128);
        gemm_bf16_kernel<<<grid, 256, 0, stream>>>(A, W, bias, C, K, N, relu);
    };

    // ---- gather ragged -> padded bf16 ----
    gather_pad_kernel<<<(RTOT * 32) / 256, 256, 0, stream>>>(nh, counts, offs, bufA);

    // ---- K / Q / V MLPs (shared-weight dense GEMMs over all atoms) ----
    gemm(bufA, wbf[3], wkB[0], bufB, RTOT, 256, 256, 1);
    gemm(bufB, wbf[4], wkB[1], bufC, RTOT, 256, 256, 1);
    gemm(bufC, wbf[5], wkB[2], bufK, RTOT, 128, 256, 0);

    gemm(bufA, wbf[0], wqB[0], bufB, RTOT, 256, 256, 1);
    gemm(bufB, wbf[1], wqB[1], bufC, RTOT, 256, 256, 1);
    gemm(bufC, wbf[2], wqB[2], bufQ, RTOT, 128, 256, 0);

    gemm(bufA, wbf[6], wvB[0], bufB, RTOT, 256, 256, 1);
    gemm(bufB, wbf[7], wvB[1], bufC, RTOT, 256, 256, 1);
    gemm(bufC, wbf[8], wvB[2], bufV, RTOT, 128, 256, 0);

    // ---- attention (corr overwrites bufA, inp is dead) ----
    bf16* corr = bufA;
    attn_kernel<<<BATCH, 256, 0, stream>>>(bufK, bufQ, bufV, counts, corr);

    // ---- uk MLP -> K2 (overwrites bufQ, Q is dead) ----
    gemm(corr, wbf[9],  ukB[0], bufB, RTOT, 256, 128, 1);
    gemm(bufB, wbf[10], ukB[1], bufC, RTOT, 256, 256, 1);
    gemm(bufC, wbf[11], ukB[2], bufQ, RTOT, 128, 256, 0);
    bf16* K2 = bufQ;

    // ---- uq MLP over batch (M=4096 GEMMs) ----
    concat_kernel<<<(BATCH * 128) / 256, 256, 0, stream>>>(dstr, tgtr, ucat);
    gemm(ucat, wbf[12], uqB[0], u1, BATCH, 384, 1024, 1);
    gemm(u1,   wbf[13], uqB[1], u2, BATCH, 256, 384,  1);
    gemm(u2,   wbf[14], uqB[2], q2, BATCH, 128, 256,  0);

    // ---- logits, sel MLP + sigmoid, ragged scatter ----
    logits_kernel<<<(BATCH * 64) / 256, 256, 0, stream>>>(K2, q2, logits);
    sel_kernel<<<BATCH, 128, 0, stream>>>(logits, selW0, selB0, selW1, selB1, selbuf);
    scatter_kernel<<<(Nnodes + 255) / 256, 256, 0, stream>>>(selbuf, nbat, npos,
                                                             (float*)d_out, Nnodes);
}
